// FlashCrossAttnWG_77558519431683
// MI455X (gfx1250) — matrix operand order repaired and checked by reading, not compiled, not run on hardware
//
#include <hip/hip_runtime.h>
#include <math.h>

// ---------------------------------------------------------------------------
// Problem constants (from reference)
// ---------------------------------------------------------------------------
#define Bsz    2
#define NQ     2048
#define NK     2048
#define QD     1024
#define CD     768
#define NH     16
#define DH     64
#define INNER  1024   // NH*DH

typedef unsigned short u16;
typedef __attribute__((ext_vector_type(16))) __bf16 v16bf;
typedef __attribute__((ext_vector_type(8)))  float  v8f;
typedef __attribute__((ext_vector_type(4)))  int    v4i;

typedef __attribute__((address_space(1))) v4i* gptr_v4i;  // global
typedef __attribute__((address_space(3))) v4i* lptr_v4i;  // LDS

// gfx1250 async direct-to-LDS path (ASYNCcnt), guarded so the file always
// compiles; fallback is global_load_b128 + ds_store_b128.
#if __has_builtin(__builtin_amdgcn_global_load_async_to_lds_b128) && \
    __has_builtin(__builtin_amdgcn_s_wait_asynccnt)
#define USE_ASYNC_LDS 1
#else
#define USE_ASYNC_LDS 0
#endif

__device__ __forceinline__ void tile_load16(void* lds_dst, const void* gsrc) {
#if USE_ASYNC_LDS
    __builtin_amdgcn_global_load_async_to_lds_b128(
        (gptr_v4i)gsrc, (lptr_v4i)lds_dst, /*offset=*/0, /*cpol=*/0);
#else
    *(uint4*)lds_dst = *(const uint4*)gsrc;
#endif
}

__device__ __forceinline__ void tile_load_fence() {
#if USE_ASYNC_LDS
    __builtin_amdgcn_s_wait_asynccnt(0);
#endif
}

union FragB16 { v16bf v; uint4 q[2]; };

__device__ __forceinline__ u16 f2bf(float x) {
    unsigned int u = __float_as_uint(x);
    unsigned int r = u + 0x7FFFu + ((u >> 16) & 1u);   // round-to-nearest-even
    return (u16)(r >> 16);
}

__device__ __forceinline__ v8f wmma_bf16(v16bf a, v16bf b, v8f c) {
    return __builtin_amdgcn_wmma_f32_16x16x32_bf16(
        /*neg_a=*/false, a, /*neg_b=*/false, b,
        /*c_mod=*/(short)0, c, /*reuse_a=*/false, /*reuse_b=*/false);
}

// A-fragment (16x32 bf16, M x K): lane L -> row m=L&15, group g=L>>4
//   elems 0..7  : K = 8g + e        (16B contiguous)
//   elems 8..15 : K = 16 + 8g + e-8 (16B contiguous)
__device__ __forceinline__ v16bf load_frag_a(const u16* lds, int row_base, int kb, int ld) {
    const int lane = threadIdx.x & 31;
    const int m = lane & 15, g = lane >> 4;
    FragB16 f;
    const u16* p = lds + (row_base + m) * ld + kb + 8 * g;
    f.q[0] = *(const uint4*)(p);
    f.q[1] = *(const uint4*)(p + 16);
    return f.v;
}

// B-fragment (32x16 bf16, K x N) from "rows hold K contiguous" storage:
// lane L -> col n=L&15, group g=L>>4; elems 0..7: K = 8g + e, elems 8..15: K = 16 + 8g + e-8
__device__ __forceinline__ v16bf load_frag_b(const u16* lds, int col_base, int kb, int ld) {
    const int lane = threadIdx.x & 31;
    const int n = lane & 15, g = lane >> 4;
    FragB16 f;
    const u16* p = lds + (col_base + n) * ld + kb + 8 * g;
    f.q[0] = *(const uint4*)(p);
    f.q[1] = *(const uint4*)(p + 16);
    return f.v;
}

// ---------------------------------------------------------------------------
// fp32 -> bf16 pack (n divisible by 4)
// ---------------------------------------------------------------------------
struct alignas(8) U16x4 { u16 s[4]; };

__global__ __launch_bounds__(256) void cvt_bf16_kernel(const float* __restrict__ src,
                                                       u16* __restrict__ dst, int n4) {
    int i = blockIdx.x * 256 + threadIdx.x;
    if (i < n4) {
        float4 v = *(const float4*)(src + (size_t)i * 4);
        U16x4 o;
        o.s[0] = f2bf(v.x); o.s[1] = f2bf(v.y); o.s[2] = f2bf(v.z); o.s[3] = f2bf(v.w);
        *(U16x4*)(dst + (size_t)i * 4) = o;
    }
}

// ---------------------------------------------------------------------------
// Tiled bf16 GEMM: out[row, col] = sum_k A[row,k] * W[col,k]
// rows = 4096 (B*seq), cols = 1024, K in {1024, 768}.
// Block tile 128x128, K-step 32; 8 waves as 2(M) x 4(N); wave tile 64x32.
// MODE: 0 = Q proj (*DH^-0.5, out [b,h,n,d])   1 = K proj (out [b,h,n,d])
//       2 = V proj (out transposed [b,h,d,n])  3 = final (fp32 + bias)
// ---------------------------------------------------------------------------
#define LDT 40   // 32 + 8 pad (bf16 elems)

template <int MODE>
__global__ __launch_bounds__(256) void gemm128_kernel(const u16* __restrict__ A,
                                                      const u16* __restrict__ W,
                                                      u16* __restrict__ outb,
                                                      float* __restrict__ outf,
                                                      const float* __restrict__ bias,
                                                      int K) {
    __shared__ __align__(16) u16 As[128 * LDT];
    __shared__ __align__(16) u16 Ws[128 * LDT];

    const int tid = threadIdx.x;
    const int wave = tid >> 5, lane = tid & 31;
    const int wm = wave >> 2, wn = wave & 3;       // 2 x 4 wave grid
    const int rbase = blockIdx.y * 128;
    const int cbase = blockIdx.x * 128;
    const int half = lane >> 4, ln = lane & 15;

    v8f acc[4][2];
#pragma unroll
    for (int mi = 0; mi < 4; ++mi)
#pragma unroll
        for (int ni = 0; ni < 2; ++ni) acc[mi][ni] = (v8f)(0.0f);

    for (int kb = 0; kb < K; kb += 32) {
        __syncthreads();
#pragma unroll
        for (int j = 0; j < 2; ++j) {
            int c = tid + j * 256;          // 512 16B chunks per array
            int row = c >> 2;
            int kc = (c & 3) * 8;
            tile_load16(As + row * LDT + kc, A + (size_t)(rbase + row) * K + kb + kc);
            tile_load16(Ws + row * LDT + kc, W + (size_t)(cbase + row) * K + kb + kc);
        }
        tile_load_fence();
        __syncthreads();

        if (kb + 32 < K) {   // hint next K tile into cache (global_prefetch_b8)
            __builtin_prefetch(A + (size_t)(rbase + (tid >> 1)) * K + kb + 32, 0, 3);
            __builtin_prefetch(W + (size_t)(cbase + (tid >> 1)) * K + kb + 32, 0, 3);
        }

        v16bf bfr[2];
#pragma unroll
        for (int ni = 0; ni < 2; ++ni)
            bfr[ni] = load_frag_b(Ws, wn * 32 + ni * 16, 0, LDT);
#pragma unroll
        for (int mi = 0; mi < 4; ++mi) {
            v16bf a = load_frag_a(As, wm * 64 + mi * 16, 0, LDT);
#pragma unroll
            for (int ni = 0; ni < 2; ++ni)
                acc[mi][ni] = wmma_bf16(a, bfr[ni], acc[mi][ni]);
        }
    }

    // Epilogue: C/D layout -> lane holds rows (half*8 + r), col = ln
#pragma unroll
    for (int mi = 0; mi < 4; ++mi)
#pragma unroll
        for (int ni = 0; ni < 2; ++ni)
#pragma unroll
            for (int r = 0; r < 8; ++r) {
                int row = rbase + wm * 64 + mi * 16 + half * 8 + r;  // [0,4096)
                int col = cbase + wn * 32 + ni * 16 + ln;            // [0,1024)
                float v = acc[mi][ni][r];
                if (MODE == 0 || MODE == 1) {
                    int b = row >> 11, n = row & 2047;
                    int h = col >> 6, d = col & 63;
                    float s = (MODE == 0) ? v * 0.125f : v;  // DH^-0.5 = 1/8
                    outb[((((size_t)b * NH + h) * 2048) + n) * DH + d] = f2bf(s);
                } else if (MODE == 2) {
                    int b = row >> 11, n = row & 2047;
                    int h = col >> 6, d = col & 63;
                    outb[(((size_t)b * NH + h) * DH + d) * 2048 + n] = f2bf(v);
                } else {
                    outf[(size_t)row * QD + col] = v + bias[col];
                }
            }
}

// ---------------------------------------------------------------------------
// Flash attention: per block = (b,h) head, 64 query rows, stream over NK.
// 8 waves as 4(M-q) x 2(N); wave S-tile 16x32, wave O-tile 16x32.
// ---------------------------------------------------------------------------
#define LDA 72    // 64 + 8 pad (bf16 elems)
#define LDM 80    // mask tile row stride (bytes), keeps 16B alignment
#define NEGBIG (-1.70141173e+38f)

__global__ __launch_bounds__(256) void attn64_kernel(const u16* __restrict__ Qh,
                                                     const u16* __restrict__ Kt,
                                                     const u16* __restrict__ Vt,
                                                     const unsigned char* __restrict__ mask,
                                                     u16* __restrict__ AO) {
    __shared__ __align__(16) u16 Qs[64 * LDA];
    __shared__ __align__(16) u16 Ks[64 * LDA];
    __shared__ __align__(16) u16 Vs[64 * LDA];
    __shared__ __align__(16) u16 Ps[64 * LDA];
    __shared__ __align__(16) unsigned char Ms[64 * LDM];
    __shared__ float m_i[64], l_i[64], rfac[64];
    __shared__ float pmax[2][64], psum[2][64];

    const int tid = threadIdx.x, wave = tid >> 5, lane = tid & 31;
    const int wm = wave >> 1, wn = wave & 1;       // 4 x 2 wave grid
    const int half = lane >> 4, ln = lane & 15;
    const int bh = blockIdx.y;
    const int b = bh >> 4, h = bh & 15;
    const int qbase = blockIdx.x * 64;

    const size_t qoff = ((size_t)bh * NQ + qbase) * DH;
    const size_t koff = (size_t)bh * NK * DH;
    const size_t voff = (size_t)bh * DH * NK;
    const size_t moff = (size_t)b * NQ * NK + (size_t)qbase * NK;

    // Load Q tile (64 x 64 bf16) -- async; fenced before first use below.
#pragma unroll
    for (int j = 0; j < 2; ++j) {
        int c = tid + j * 256;
        int row = c >> 3, kc = (c & 7) * 8;
        tile_load16(Qs + row * LDA + kc, Qh + qoff + (size_t)row * DH + kc);
    }
    if (tid < 64) { m_i[tid] = -INFINITY; l_i[tid] = 0.0f; }

    v8f o[2];
    o[0] = (v8f)(0.0f); o[1] = (v8f)(0.0f);

    for (int kt = 0; kt < NK; kt += 64) {
        __syncthreads();   // protect Ks/Vs/Ps/Ms reuse
        // K tile [64 keys x 64 d], V tile [64 d x 64 keys], mask tile [64 q x 64 k]
#pragma unroll
        for (int j = 0; j < 2; ++j) {
            int c = tid + j * 256;
            int row = c >> 3, kc = (c & 7) * 8;
            tile_load16(Ks + row * LDA + kc, Kt + koff + (size_t)(kt + row) * DH + kc);
            tile_load16(Vs + row * LDA + kc, Vt + voff + (size_t)row * NK + kt + kc);
        }
        {
            int row = tid >> 2, kc = (tid & 3) * 16;
            tile_load16(Ms + row * LDM + kc, mask + moff + (size_t)row * NK + kt + kc);
        }
        tile_load_fence();
        __syncthreads();

        // ---- S = Q K^T (scale already folded into Q) ----
        v8f s[2];
        s[0] = (v8f)(0.0f); s[1] = (v8f)(0.0f);
#pragma unroll
        for (int ks = 0; ks < DH; ks += 32) {
            v16bf a = load_frag_a(Qs, wm * 16, ks, LDA);
#pragma unroll
            for (int ni = 0; ni < 2; ++ni) {
                v16bf bb = load_frag_b(Ks, wn * 32 + ni * 16, ks, LDA);
                s[ni] = wmma_bf16(a, bb, s[ni]);
            }
        }

        // ---- additive mask bias (from LDS tile) ----
#pragma unroll
        for (int ni = 0; ni < 2; ++ni)
#pragma unroll
            for (int r = 0; r < 8; ++r) {
                int rq = wm * 16 + half * 8 + r;
                int rk = wn * 32 + ni * 16 + ln;
                if (!Ms[rq * LDM + rk]) s[ni][r] += NEGBIG;
            }

        // ---- partial row max over this wave's 32 cols ----
        float v[8];
#pragma unroll
        for (int r = 0; r < 8; ++r) v[r] = fmaxf(s[0][r], s[1][r]);
#pragma unroll
        for (int off = 1; off < 16; off <<= 1)
#pragma unroll
            for (int r = 0; r < 8; ++r) v[r] = fmaxf(v[r], __shfl_xor(v[r], off, 32));
        if (ln == 0)
#pragma unroll
            for (int r = 0; r < 8; ++r) pmax[wn][wm * 16 + half * 8 + r] = v[r];
        __syncthreads();

        if (tid < 64) {
            float nm = fmaxf(m_i[tid], fmaxf(pmax[0][tid], pmax[1][tid]));
            float rf = __expf(m_i[tid] - nm);
            rfac[tid] = rf;
            l_i[tid] *= rf;
            m_i[tid] = nm;
        }
        __syncthreads();

        // ---- exp, P -> LDS (bf16), partial row-sum, rescale O ----
#pragma unroll
        for (int r = 0; r < 8; ++r) {
            int row = wm * 16 + half * 8 + r;
            float nm = m_i[row];
            float p0 = __expf(s[0][r] - nm);
            float p1 = __expf(s[1][r] - nm);
            Ps[row * LDA + wn * 32 + ln]      = f2bf(p0);
            Ps[row * LDA + wn * 32 + 16 + ln] = f2bf(p1);
            v[r] = p0 + p1;
            float rf = rfac[row];
            o[0][r] *= rf;
            o[1][r] *= rf;
        }
#pragma unroll
        for (int off = 1; off < 16; off <<= 1)
#pragma unroll
            for (int r = 0; r < 8; ++r) v[r] += __shfl_xor(v[r], off, 32);
        if (ln == 0)
#pragma unroll
            for (int r = 0; r < 8; ++r) psum[wn][wm * 16 + half * 8 + r] = v[r];
        __syncthreads();

        if (tid < 64) l_i[tid] += psum[0][tid] + psum[1][tid];

        // ---- O += P V ----
#pragma unroll
        for (int ks = 0; ks < 64; ks += 32) {
            v16bf a = load_frag_a(Ps, wm * 16, ks, LDA);
#pragma unroll
            for (int ni = 0; ni < 2; ++ni) {
                v16bf bb = load_frag_b(Vs, wn * 32 + ni * 16, ks, LDA);
                o[ni] = wmma_bf16(a, bb, o[ni]);
            }
        }
    }
    __syncthreads();

    // ---- finalize: O / l_i -> AO [b, q, h*64 + d] (bf16) ----
#pragma unroll
    for (int ni = 0; ni < 2; ++ni)
#pragma unroll
        for (int r = 0; r < 8; ++r) {
            int row = wm * 16 + half * 8 + r;
            float inv = 1.0f / l_i[row];
            int q = qbase + row;
            int dcol = wn * 32 + ni * 16 + ln;
            AO[((size_t)b * NQ + q) * INNER + h * DH + dcol] = f2bf(o[ni][r] * inv);
        }
}

// ---------------------------------------------------------------------------
// Host launcher
// ---------------------------------------------------------------------------
extern "C" void kernel_launch(void* const* d_in, const int* in_sizes, int n_in,
                              void* d_out, int out_size, void* d_ws, size_t ws_size,
                              hipStream_t stream) {
    const float* x   = (const float*)d_in[0];            // [B,NQ,QD]
    const float* ctx = (const float*)d_in[1];            // [B,NK,CD]
    const unsigned char* mask = (const unsigned char*)d_in[2]; // [B,1,NQ,NK] bool
    const float* Wq  = (const float*)d_in[3];            // [INNER,QD]
    const float* Wk  = (const float*)d_in[4];            // [INNER,CD]
    const float* Wv  = (const float*)d_in[5];            // [INNER,CD]
    const float* Wo  = (const float*)d_in[6];            // [QD,INNER]
    const float* bo  = (const float*)d_in[7];            // [QD]
    float* out = (float*)d_out;                          // [B,NQ,QD] fp32

    // Workspace layout (bf16 elements)
    u16* ws = (u16*)d_ws;
    const size_t SZ_XB = (size_t)Bsz * NQ * QD;      // 4,194,304
    const size_t SZ_CB = (size_t)Bsz * NK * CD;      // 3,145,728
    const size_t SZ_WQ = (size_t)INNER * QD;
    const size_t SZ_WK = (size_t)INNER * CD;
    const size_t SZ_WV = (size_t)INNER * CD;
    const size_t SZ_WO = (size_t)QD * INNER;
    const size_t SZ_H  = (size_t)Bsz * NH * 2048 * DH;

    u16* xb  = ws;                 size_t off = SZ_XB;
    u16* cb  = ws + off;           off += SZ_CB;
    u16* wqb = ws + off;           off += SZ_WQ;
    u16* wkb = ws + off;           off += SZ_WK;
    u16* wvb = ws + off;           off += SZ_WV;
    u16* wob = ws + off;           off += SZ_WO;
    u16* Qh  = ws + off;           off += SZ_H;     // [b,h,nq,d]
    u16* Kt  = ws + off;           off += SZ_H;     // [b,h,nk,d]
    u16* Vt  = ws + off;           off += SZ_H;     // [b,h,d,nk]
    u16* AO  = ws + off;           off += SZ_XB;    // [b,nq,inner]

    auto cvt = [&](const float* s, u16* d, size_t n) {
        int n4 = (int)(n / 4);
        cvt_bf16_kernel<<<(n4 + 255) / 256, 256, 0, stream>>>(s, d, n4);
    };
    cvt(x,   xb,  SZ_XB);
    cvt(ctx, cb,  SZ_CB);
    cvt(Wq,  wqb, SZ_WQ);
    cvt(Wk,  wkb, SZ_WK);
    cvt(Wv,  wvb, SZ_WV);
    cvt(Wo,  wob, SZ_WO);

    dim3 ggrid(INNER / 128, (Bsz * 2048) / 128);   // (8, 32)
    gemm128_kernel<0><<<ggrid, 256, 0, stream>>>(xb, wqb, Qh, nullptr, nullptr, QD);
    gemm128_kernel<1><<<ggrid, 256, 0, stream>>>(cb, wkb, Kt, nullptr, nullptr, CD);
    gemm128_kernel<2><<<ggrid, 256, 0, stream>>>(cb, wvb, Vt, nullptr, nullptr, CD);

    dim3 agrid(NQ / 64, Bsz * NH);                 // (32, 32)
    attn64_kernel<<<agrid, 256, 0, stream>>>(Qh, Kt, Vt, mask, AO);

    gemm128_kernel<3><<<ggrid, 256, 0, stream>>>(AO, wob, nullptr, out, bo, INNER);
}
